// Attention_15436112462268
// MI455X (gfx1250) — matrix operand order repaired and checked by reading, not compiled, not run on hardware
//
#include <hip/hip_runtime.h>
#include <math.h>

typedef _Float16 half_t;
typedef __attribute__((ext_vector_type(16))) _Float16 v16h;
typedef __attribute__((ext_vector_type(8)))  _Float16 v8h;
typedef __attribute__((ext_vector_type(8)))  float    v8f;

#define LDS_STRIDE 520   // halves; 1040B row stride -> bank shift of 4/row, conflict-free

// ---------------- weight conversion kernels ----------------
// Row-permute Wq/Wk/Wv so GEMM output columns come out head-major:
// output col o' = h*128+d  <->  original row o = d*4+h.
__global__ __launch_bounds__(512) void cvt_w_rowperm(const float* __restrict__ W,
                                                     half_t* __restrict__ out) {
  int op = blockIdx.x;                 // o' (head-major), 512
  int c  = threadIdx.x;                // k, 512
  int dd = op & 127, h = op >> 7;
  int o  = dd * 4 + h;                 // original output index
  out[op * 512 + c] = (half_t)W[o * 512 + c];
}

// Column-permute Wo from interleaved o = d*4+h to head-major o' = h*128+d,
// matching the layout y is produced in.
__global__ __launch_bounds__(512) void cvt_wo_perm(const float* __restrict__ Wo,
                                                   half_t* __restrict__ out) {
  int oo = blockIdx.x;                 // output row, 512
  int o  = threadIdx.x;                // input col, 512
  int op = (o & 3) * 128 + (o >> 2);
  out[oo * 512 + op] = (half_t)Wo[oo * 512 + o];
}

// ---------------- fused f32->f16 + QKV GEMM + bias + scale + RoPE ----------------
// Grid: N/16 blocks, 256 threads (8 waves). Tiles 0..31 -> Q, 32..63 -> K, 64..95 -> V.
// Output columns are head-major o' (row-permuted weights), so stores are coalesced
// and the RoPE partner is lane^1.
__global__ __launch_bounds__(256) void qkv_rope(
    const float* __restrict__ x, const half_t* __restrict__ Whp,
    const float* __restrict__ bq, const float* __restrict__ bk,
    const float* __restrict__ bv, const float* __restrict__ pos,
    half_t* __restrict__ qb, half_t* __restrict__ kb, half_t* __restrict__ vb)
{
  __shared__ __align__(16) half_t xs[16 * LDS_STRIDE];
  __shared__ float pos_s[48];
  const int tid = threadIdx.x;
  const int node0 = blockIdx.x * 16;

  // stage 16x512 tile: read fp32 x, convert to f16 in LDS (no xh round-trip)
  for (int i = tid; i < 1024; i += 256) {
    int r = i >> 6, c8 = (i & 63) << 3;
    const float* src = &x[(size_t)(node0 + r) * 512 + c8];
    v8h t;
    #pragma unroll
    for (int e = 0; e < 8; ++e) t[e] = (half_t)src[e];
    *(v8h*)&xs[r * LDS_STRIDE + c8] = t;
  }
  if (tid < 48) pos_s[tid] = pos[node0 * 3 + tid];
  __syncthreads();

  const int lane = tid & 31, wave = tid >> 5;
  const int n = lane & 15, hb = lane >> 4;

  for (int tile = wave; tile < 96; tile += 8) {
    const int mat = tile >> 5;               // 0=q 1=k 2=v
    const int opb = (tile & 31) << 4;        // o' base (head-major)
    const int oP  = opb + n;                 // o' = h*128 + dd
    const half_t* Wrow = Whp + (size_t)mat * 262144 + (size_t)oP * 512;

    v8f c = {};
    for (int ks = 0; ks < 512; ks += 32) {
      v16h a, b;
      // A (16x32 f16): lane holds row m=n; lane-half selects K 0-15 / 16-31
      *((v8h*)&a)     = *(const v8h*)&xs[n * LDS_STRIDE + ks + hb * 16];
      *((v8h*)&a + 1) = *(const v8h*)&xs[n * LDS_STRIDE + ks + hb * 16 + 8];
      // B (32x16 f16): lane = col N=n, lane-half selects K 0-15 / 16-31
      b = *(const v16h*)&Wrow[ks + hb * 16];
      c = __builtin_amdgcn_wmma_f32_16x16x32_f16(false, a, false, b, (short)0, c,
                                                 false, false);
    }

    const int dd = oP & 127, h = oP >> 7;
    const float* bsel = (mat == 0) ? bq : (mat == 1) ? bk : bv;
    const float bias  = bsel[dd * 4 + h];
    const float scale = (mat == 0) ? 0.08838834764831845f : 1.0f; // 1/sqrt(128)
    const bool doRope = (mat < 2) && (dd < 126);
    float invf = 0.f, esign = 0.f;
    int p = 0;
    if (doRope) {
      p = dd / 42;                           // POS_DIM index
      int m2 = (dd % 42) >> 1;               // freq index
      invf = __expf(-0.43858764f * (float)m2);  // ln(1e4)/21
      esign = (dd & 1) ? 1.0f : -1.0f;       // even: -sin*partner; odd: +sin*partner
    }
    half_t* dst = (mat == 0) ? qb : (mat == 1) ? kb : vb;

    #pragma unroll
    for (int r = 0; r < 8; ++r) {
      // C layout: lane holds N fixed; M = r + 8*hb
      float val = (c[r] + bias) * scale;
      float partner = __shfl_xor(val, 1, 32);   // pair: dd ^ 1 <=> o' ^ 1 <=> lane ^ 1
      if (doRope) {
        int row = r + hb * 8;
        float ang = pos_s[row * 3 + p] * invf;
        float sn, cs;
        __sincosf(ang, &sn, &cs);
        val = val * cs + partner * (esign * sn);
      }
      dst[(size_t)(node0 + r + hb * 8) * 512 + oP] = (half_t)val;   // coalesced
    }
  }
}

// ---------------- fused edge attention + output GEMM ----------------
// Grid: N/16 blocks, 256 threads (8 waves) = 16 nodes x 16 threads.
// Phase 1: scores (thread = (node, edge), 4 heads), softmax over 16 edges.
// Phase 2: y accumulation into registers -> f16 LDS tile.
// Phase 3: 32 WMMA N-tiles of y @ Wo_perm.T + bo -> fp32 out. No y round-trip.
__global__ __launch_bounds__(256) void attn_out(
    const half_t* __restrict__ qb, const half_t* __restrict__ kb,
    const half_t* __restrict__ vb, const int* __restrict__ cols,
    const half_t* __restrict__ Woh, const float* __restrict__ bo,
    float* __restrict__ out)
{
  __shared__ __align__(16) half_t qs[16 * 512];         // 16 KB
  __shared__ __align__(16) half_t ys[16 * LDS_STRIDE];  // 16.6 KB
  __shared__ float sc[16][16][4];
  __shared__ float at[16][16][4];
  __shared__ int   cidx[16][16];
  const int tid = threadIdx.x;
  const int node0 = blockIdx.x * 16;

  // stage q (head-major f16) for 16 nodes
  for (int i = tid; i < 1024; i += 256) {
    int r = i >> 6, c8 = (i & 63) << 3;
    *(v8h*)&qs[r * 512 + c8] = *(const v8h*)&qb[(size_t)(node0 + r) * 512 + c8];
  }
  cidx[tid >> 4][tid & 15] = cols[node0 * 16 + tid];    // 256 edges
  __syncthreads();

  const int nl = tid >> 4;          // node-local 0..15
  const int t16 = tid & 15;
  const int cj = cidx[nl][t16];     // this thread's edge

  // ---- scores: dot(q[node], k[cj]) per head (head-major: contiguous 256B/head)
  {
    const v8h* kp = (const v8h*)&kb[(size_t)cj * 512];
    const v8h* qp = (const v8h*)&qs[nl * 512];
    #pragma unroll
    for (int h = 0; h < 4; ++h) {
      float s = 0.f;
      #pragma unroll 4
      for (int d8 = 0; d8 < 16; ++d8) {
        v8h kk = kp[h * 16 + d8], qq = qp[h * 16 + d8];
        #pragma unroll
        for (int r = 0; r < 8; ++r) s += (float)qq[r] * (float)kk[r];
      }
      sc[nl][t16][h] = s;
    }
  }
  __syncthreads();

  // ---- softmax over the 16 edges of this node, per head
  #pragma unroll
  for (int h = 0; h < 4; ++h) {
    float mx = -3.4e38f;
    for (int jj = 0; jj < 16; ++jj) mx = fmaxf(mx, sc[nl][jj][h]);
    float sum = 0.f;
    for (int jj = 0; jj < 16; ++jj) sum += __expf(sc[nl][jj][h] - mx);
    at[nl][t16][h] = __expf(sc[nl][t16][h] - mx) / sum;
  }
  __syncthreads();

  // ---- y accumulation: thread owns 32 contiguous head-major dims of its node
  {
    const int ob = t16 * 32;            // o' base
    const int hseg = t16 >> 2;          // head of this 32-dim slice
    float acc[32] = {};
    for (int jj = 0; jj < 16; ++jj) {
      const float a = at[nl][jj][hseg];
      const v8h* vp = (const v8h*)&vb[(size_t)cidx[nl][jj] * 512 + ob];
      #pragma unroll
      for (int q8 = 0; q8 < 4; ++q8) {
        v8h vv = vp[q8];
        #pragma unroll
        for (int r = 0; r < 8; ++r) acc[q8 * 8 + r] += a * (float)vv[r];
      }
    }
    #pragma unroll
    for (int q8 = 0; q8 < 4; ++q8) {
      v8h t;
      #pragma unroll
      for (int r = 0; r < 8; ++r) t[r] = (half_t)acc[q8 * 8 + r];
      *(v8h*)&ys[nl * LDS_STRIDE + ob + q8 * 8] = t;
    }
  }
  __syncthreads();

  // ---- output GEMM: out = ys @ Woh.T + bo (fp32)
  const int lane = tid & 31, wave = tid >> 5;
  const int n = lane & 15, hb = lane >> 4;
  for (int tile = wave; tile < 32; tile += 8) {
    const int o = tile * 16 + n;                  // original output ordering
    const half_t* Wrow = Woh + (size_t)o * 512;   // columns already head-major
    v8f c = {};
    for (int ks = 0; ks < 512; ks += 32) {
      v16h a, b;
      *((v8h*)&a)     = *(const v8h*)&ys[n * LDS_STRIDE + ks + hb * 16];
      *((v8h*)&a + 1) = *(const v8h*)&ys[n * LDS_STRIDE + ks + hb * 16 + 8];
      b = *(const v16h*)&Wrow[ks + hb * 16];
      c = __builtin_amdgcn_wmma_f32_16x16x32_f16(false, a, false, b, (short)0, c,
                                                 false, false);
    }
    const float bias = bo[o];
    #pragma unroll
    for (int r = 0; r < 8; ++r)
      out[(size_t)(node0 + r + hb * 8) * 512 + o] = c[r] + bias;
  }
}

// ---------------- launcher ----------------
extern "C" void kernel_launch(void* const* d_in, const int* in_sizes, int n_in,
                              void* d_out, int out_size, void* d_ws, size_t ws_size,
                              hipStream_t stream)
{
  (void)n_in; (void)out_size; (void)ws_size;
  const float* x   = (const float*)d_in[0];
  const float* pos = (const float*)d_in[1];
  /* d_in[2] rows: implicit (deg=16, sorted) */
  const int*   cols= (const int*)  d_in[3];
  const float* Wq  = (const float*)d_in[4];
  const float* bq  = (const float*)d_in[5];
  const float* Wk  = (const float*)d_in[6];
  const float* bk  = (const float*)d_in[7];
  const float* Wv  = (const float*)d_in[8];
  const float* bv  = (const float*)d_in[9];
  const float* Wo  = (const float*)d_in[10];
  const float* bo  = (const float*)d_in[11];
  float* out = (float*)d_out;

  const int N = in_sizes[0] / 512;        // 20000 (multiple of 16)
  const size_t nhBytes = (size_t)N * 512 * sizeof(half_t);

  char* ws = (char*)d_ws;
  size_t off = 0;
  half_t* Whp = (half_t*)(ws + off); off += 3 * 262144 * sizeof(half_t);
  half_t* Woh = (half_t*)(ws + off); off += 262144 * sizeof(half_t);
  half_t* qb  = (half_t*)(ws + off); off += nhBytes;
  half_t* kb  = (half_t*)(ws + off); off += nhBytes;
  half_t* vb  = (half_t*)(ws + off); off += nhBytes;

  cvt_w_rowperm<<<512, 512, 0, stream>>>(Wq, Whp);
  cvt_w_rowperm<<<512, 512, 0, stream>>>(Wk, Whp + 262144);
  cvt_w_rowperm<<<512, 512, 0, stream>>>(Wv, Whp + 524288);
  cvt_wo_perm  <<<512, 512, 0, stream>>>(Wo, Woh);

  qkv_rope<<<N / 16, 256, 0, stream>>>(x, Whp, bq, bk, bv, pos, qb, kb, vb);
  attn_out<<<N / 16, 256, 0, stream>>>(qb, kb, vb, cols, Woh, bo, out);
}
